// Head_63634235457862
// MI455X (gfx1250) — matrix operand order repaired and checked by reading, not compiled, not run on hardware
//
#include <hip/hip_runtime.h>

typedef __attribute__((ext_vector_type(16))) _Float16 v16h;
typedef __attribute__((ext_vector_type(8)))  _Float16 v8h;
typedef __attribute__((ext_vector_type(8)))  float    v8f;
typedef __attribute__((ext_vector_type(4)))  float    v4f;

#define T_DIM 128
#define C_DIM 256
#define H_DIM 64

// ---- WMMA fragment loaders (ISA 7.12.2 layouts, wave32) --------------------
// A (16x32 f16): lanes 0-15 -> row M=l16, halves {kb..kb+7, kb+16..kb+23}
//                lanes 16-31 -> row M=l16, halves {kb+8..kb+15, kb+24..kb+31}
__device__ __forceinline__ v16h load_a_f16(const _Float16* rowbase, int kb, int sel) {
  const v8h* p0 = (const v8h*)(rowbase + kb + 8 * sel);
  const v8h* p1 = (const v8h*)(rowbase + kb + 16 + 8 * sel);
  v8h c0 = *p0, c1 = *p1;
  v16h a;
#pragma unroll
  for (int j = 0; j < 8; ++j) { a[j] = c0[j]; a[8 + j] = c1[j]; }
  return a;
}

// Same A layout but source rows are fp32 in global memory (converted to f16).
__device__ __forceinline__ v16h load_a_f32cvt(const float* rowbase, int kb, int sel) {
  const v4f* p0 = (const v4f*)(rowbase + kb + 8 * sel);
  const v4f* p1 = (const v4f*)(rowbase + kb + 16 + 8 * sel);
  v4f f0 = p0[0], f1 = p0[1], f2 = p1[0], f3 = p1[1];
  v16h a;
#pragma unroll
  for (int j = 0; j < 4; ++j) {
    a[j]      = (_Float16)f0[j];
    a[4 + j]  = (_Float16)f1[j];
    a[8 + j]  = (_Float16)f2[j];
    a[12 + j] = (_Float16)f3[j];
  }
  return a;
}

// B (32x16 f16): lane holds column n=l16; k-halves {kb+8*sel..+7, kb+16+8*sel..+7}
// (same k order as A). rowbase must point at the (transposed) row for column n.
__device__ __forceinline__ v16h load_b_f16(const _Float16* rowbase, int kb, int sel) {
  const v8h* p0 = (const v8h*)(rowbase + kb + 8 * sel);
  const v8h* p1 = (const v8h*)(rowbase + kb + 16 + 8 * sel);
  v8h c0 = *p0, c1 = *p1;
  v16h bv;
#pragma unroll
  for (int j = 0; j < 8; ++j) { bv[j] = c0[j]; bv[8 + j] = c1[j]; }
  return bv;
}

// ---- pre-kernel: Wq/Wk/Wv fp32 [C][H] -> f16 transposed [H][C] in ws -------
__global__ __launch_bounds__(256) void convert_w(const float* __restrict__ Wk,
                                                 const float* __restrict__ Wq,
                                                 const float* __restrict__ Wv,
                                                 _Float16* __restrict__ wt) {
  int i = blockIdx.x * blockDim.x + threadIdx.x;
  if (i >= 3 * H_DIM * C_DIM) return;
  int mat = i / (H_DIM * C_DIM);
  int rem = i % (H_DIM * C_DIM);
  int h = rem / C_DIM;
  int c = rem % C_DIM;
  const float* W = (mat == 0) ? Wq : (mat == 1) ? Wk : Wv;  // order: Q,K,V
  wt[i] = (_Float16)W[c * H_DIM + h];
}

// ---- main kernel: one workgroup per batch element --------------------------
// LDS layout (bytes):
//   sQ  f16[128][64]   @ 0       (16 KB)
//   sK  f16[128][64]   @ 16384   (16 KB)
//   sVt f16[64][128]   @ 32768   (16 KB)
//   sS  f32[128][128]  @ 49152   (64 KB)  -- reused as P f16[128] per row
__global__ __launch_bounds__(256) void head_attn(const float* __restrict__ x,
                                                 const _Float16* __restrict__ wt,
                                                 float* __restrict__ out) {
  extern __shared__ char smem[];
  _Float16* sQ  = (_Float16*)smem;
  _Float16* sK  = (_Float16*)(smem + 16384);
  _Float16* sVt = (_Float16*)(smem + 32768);
  float*    sS  = (float*)(smem + 49152);

  const int b    = blockIdx.x;
  const int lane = threadIdx.x & 31;
  const int wave = threadIdx.x >> 5;
  const int l16  = lane & 15;
  const int sel  = lane >> 4;

  const float* xb = x + (size_t)b * (T_DIM * C_DIM);

  // ---------------- Phase 1: Q,K,V = X @ W ----------------------------------
  // wave = Mtile. For each weight matrix, each A fragment of x is loaded once
  // from global and shared across all 4 N-tiles (4 accumulator chains).
  {
    const int mt = wave;
    const float* arow = xb + (mt * 16 + l16) * C_DIM;
    for (int mat = 0; mat < 3; ++mat) {
      const _Float16* wbase = wt + mat * (H_DIM * C_DIM);
      v8f acc[4] = {{}, {}, {}, {}};
#pragma unroll
      for (int ks = 0; ks < 8; ++ks) {
        v16h a = load_a_f32cvt(arow, ks * 32, sel);
#pragma unroll
        for (int nt = 0; nt < 4; ++nt) {
          v16h bf = load_b_f16(wbase + (nt * 16 + l16) * C_DIM, ks * 32, sel);
          acc[nt] = __builtin_amdgcn_wmma_f32_16x16x32_f16(false, a, false, bf,
                                                           (short)0, acc[nt],
                                                           false, false);
        }
      }
      if (mat == 2) {  // V stored transposed: sVt[h][t]
#pragma unroll
        for (int nt = 0; nt < 4; ++nt)
#pragma unroll
          for (int r = 0; r < 8; ++r) {
            int row = mt * 16 + r + 8 * sel;
            int col = nt * 16 + l16;
            sVt[col * T_DIM + row] = (_Float16)acc[nt][r];
          }
      } else {
        _Float16* dst = (mat == 0) ? sQ : sK;
#pragma unroll
        for (int nt = 0; nt < 4; ++nt)
#pragma unroll
          for (int r = 0; r < 8; ++r) {
            int row = mt * 16 + r + 8 * sel;
            int col = nt * 16 + l16;
            dst[row * H_DIM + col] = (_Float16)acc[nt][r];
          }
      }
    }
  }
  __syncthreads();

  // ---------------- Phase 2: S = (Q @ K^T) * scale, causal tiles only -------
  const float scale = 0.125f;  // 1/sqrt(64)
  for (int idx = wave; idx < 36; idx += 8) {
    int mt = 0;
    while (((mt + 1) * (mt + 2)) / 2 <= idx) ++mt;
    int nt = idx - (mt * (mt + 1)) / 2;
    const _Float16* qrow = sQ + (mt * 16 + l16) * H_DIM;
    const _Float16* krow = sK + (nt * 16 + l16) * H_DIM;  // B = K^T
    v8f acc = {};
#pragma unroll
    for (int ks = 0; ks < 2; ++ks) {
      v16h a  = load_a_f16(qrow, ks * 32, sel);
      v16h bf = load_b_f16(krow, ks * 32, sel);
      acc = __builtin_amdgcn_wmma_f32_16x16x32_f16(false, a, false, bf,
                                                   (short)0, acc, false, false);
    }
#pragma unroll
    for (int r = 0; r < 8; ++r) {
      int row = mt * 16 + r + 8 * sel;
      sS[row * T_DIM + nt * 16 + l16] = acc[r] * scale;
    }
  }
  __syncthreads();

  // ---------------- Phase 3: causal softmax; write P f16 in place -----------
  for (int rr = 0; rr < 16; ++rr) {
    int t = wave * 16 + rr;
    float* row = sS + t * T_DIM;
    float v0[4];
    float m = -__builtin_inff();
#pragma unroll
    for (int j = 0; j < 4; ++j) {
      int c = lane + 32 * j;
      v0[j] = (c <= t) ? row[c] : -__builtin_inff();
      m = fmaxf(m, v0[j]);
    }
#pragma unroll
    for (int off = 16; off > 0; off >>= 1) m = fmaxf(m, __shfl_xor(m, off, 32));
    float e[4];
    float s = 0.f;
#pragma unroll
    for (int j = 0; j < 4; ++j) {
      int c = lane + 32 * j;
      e[j] = (c <= t) ? __expf(v0[j] - m) : 0.f;
      s += e[j];
    }
#pragma unroll
    for (int off = 16; off > 0; off >>= 1) s += __shfl_xor(s, off, 32);
    float inv = 1.0f / s;
    _Float16* prow = (_Float16*)row;  // P row: 128 halves at row base
#pragma unroll
    for (int j = 0; j < 4; ++j) {
      int c = lane + 32 * j;
      prow[c] = (_Float16)(e[j] * inv);  // zeros beyond diagonal
    }
  }
  __syncthreads();

  // ---------------- Phase 4: O = P @ V --------------------------------------
  // wave = Mtile; 4 Ntiles share each A fragment; skip all-zero k-steps.
  {
    float* ob = out + (size_t)b * (T_DIM * H_DIM);
    const _Float16* P = (const _Float16*)sS;  // row stride 256 halves
    const int mt = wave;
    const int ksteps = mt / 2 + 1;  // causality: cols <= mt*16+15
    const _Float16* prow = P + (mt * 16 + l16) * 256;
    v8f acc[4] = {{}, {}, {}, {}};
    for (int ks = 0; ks < ksteps; ++ks) {
      v16h a = load_a_f16(prow, ks * 32, sel);
#pragma unroll
      for (int nt = 0; nt < 4; ++nt) {
        v16h bf = load_b_f16(sVt + (nt * 16 + l16) * T_DIM, ks * 32, sel);
        acc[nt] = __builtin_amdgcn_wmma_f32_16x16x32_f16(false, a, false, bf,
                                                         (short)0, acc[nt],
                                                         false, false);
      }
    }
#pragma unroll
    for (int nt = 0; nt < 4; ++nt) {
#pragma unroll
      for (int r = 0; r < 8; ++r) {
        int row = mt * 16 + r + 8 * sel;
        ob[row * H_DIM + nt * 16 + l16] = acc[nt][r];
      }
    }
  }
}

extern "C" void kernel_launch(void* const* d_in, const int* in_sizes, int n_in,
                              void* d_out, int out_size, void* d_ws, size_t ws_size,
                              hipStream_t stream) {
  (void)in_sizes; (void)n_in; (void)out_size; (void)ws_size;
  const float* x  = (const float*)d_in[0];
  const float* Wk = (const float*)d_in[1];
  const float* Wq = (const float*)d_in[2];
  const float* Wv = (const float*)d_in[3];
  float* out = (float*)d_out;
  _Float16* wt = (_Float16*)d_ws;  // 3 * 64 * 256 f16 = 96 KB

  convert_w<<<(3 * H_DIM * C_DIM + 255) / 256, 256, 0, stream>>>(Wk, Wq, Wv, wt);
  head_attn<<<2048, 256, 114688, stream>>>(x, wt, out);  // 112 KB dynamic LDS
}
